// SimpleFFN_40664750358845
// MI455X (gfx1250) — matrix operand order repaired and checked by reading, not compiled, not run on hardware
//
#include <hip/hip_runtime.h>
#include <hip/hip_bf16.h>

typedef __attribute__((ext_vector_type(16))) _Float16 v16h;
typedef __attribute__((ext_vector_type(8)))  _Float16 v8h;
typedef __attribute__((ext_vector_type(8)))  float    v8f;

#define DIMC 192
#define FFNC 384
#define HW   36864        // 192*192 pixels per image
#define NT   128          // pixel tile per block
#define NSUB (NT / 16)    // 8 N-subtiles
#define SROW 200          // padded LDS row stride (halves): 400B rows -> 16B-aligned frags
#define KF   6            // 192 / 32 K-fragments
#define MT1  24           // 384/16 output tiles, GEMM1
#define MT2  12           // 192/16 output tiles, GEMM2
#define NWAVE 12          // one wave per M-tile
#define TPB  (NWAVE * 32)

// ---------------------------------------------------------------------------
// Pack f32 row-major weight matrix [Mtiles*16 x 192] into CDNA5 WMMA A-fragment
// layout (16x32 f16 per fragment): lane holds row M = mt*16 + (lane&15);
// half h maps to K = kf*32 + h + 8*((h>>3) + (lane>>4)).
// Fragment f = mt*KF + kf stored as [f][lane][16 halves] -> 32B per lane.
// ---------------------------------------------------------------------------
__global__ void pack_weights_kernel(const float* __restrict__ W,
                                    _Float16* __restrict__ dst,
                                    int Mtiles) {
  int t = blockIdx.x * blockDim.x + threadIdx.x;
  int total = Mtiles * KF * 32;
  if (t >= total) return;
  int lane = t & 31;
  int frag = t >> 5;
  int kf = frag % KF;
  int mt = frag / KF;
  int M  = mt * 16 + (lane & 15);
  int hi = lane >> 4;
  v16h v;
#pragma unroll
  for (int h = 0; h < 16; ++h) {
    int K = kf * 32 + h + 8 * ((h >> 3) + hi);
    v[h] = (_Float16)W[M * DIMC + K];
  }
  ((v16h*)dst)[frag * 32 + lane] = v;
}

__device__ __forceinline__ v16h lds_frag(const _Float16* p) {
  // two 16B-aligned LDS b128 loads
  v8h lo = *(const v8h*)p;
  v8h hi = *(const v8h*)(p + 16);
  return __builtin_shufflevector(lo, hi, 0,1,2,3,4,5,6,7,8,9,10,11,12,13,14,15);
}

// ---------------------------------------------------------------------------
// Fused FFN: x[b,c,h,w] -> gemm1(+bias) -> gate -> gemm2(+bias) -> out
// One block = 128-pixel tile of one image; 12 waves, wave w owns M-tile w.
// Weight fragments register-resident; B fragments double-buffered from LDS.
// ---------------------------------------------------------------------------
__launch_bounds__(TPB)
__global__ void simpleffn_kernel(const float* __restrict__ x,
                                 const _Float16* __restrict__ wA1,  // [24][6][32][16]
                                 const float* __restrict__ b_in,
                                 const _Float16* __restrict__ wA2,  // [12][6][32][16]
                                 const float* __restrict__ b_out,
                                 float* __restrict__ out) {
  extern __shared__ _Float16 smem[];
  _Float16* xs = smem;               // x tile, f16, pixel-major [p][c]   (NT*SROW)
  _Float16* gs = smem + NT * SROW;   // gated tile, f16, pixel-major [p][c]

  const int tid  = threadIdx.x;
  const int lane = tid & 31;
  const int wv   = tid >> 5;           // wave 0..11 == M-tile index
  const int hi   = lane >> 4;          // half-wave
  const int ln   = lane & 15;

  const int tile = blockIdx.x;         // 0..2303
  const int img  = tile / (HW / NT);
  const int pt0  = (tile % (HW / NT)) * NT;

  const float* xb = x   + (size_t)img * DIMC * HW;
  float*       ob = out + (size_t)img * DIMC * HW;
  const v16h*  A1 = (const v16h*)wA1;
  const v16h*  A2 = (const v16h*)wA2;

  // ---- GEMM1 weight fragments + biases, register-resident -----------------
  v16h a1[KF], a2[KF];
#pragma unroll
  for (int kf = 0; kf < KF; ++kf) {
    a1[kf] = A1[(wv * KF + kf) * 32 + lane];          // W_in rows [wv*16, +16)
    a2[kf] = A1[((wv + 12) * KF + kf) * 32 + lane];   // W_in rows [wv*16+192, +16)
  }
  v8f bias1, bias2;
#pragma unroll
  for (int r = 0; r < 8; ++r) {
    int o = wv * 16 + r + 8 * hi;
    bias1[r] = b_in[o];
    bias2[r] = b_in[o + DIMC];
  }

  // ---- Stage 1: global f32 -> LDS f16, transposed to pixel-major ----------
  // Each thread packs 2 consecutive channels -> one b32 LDS store.
  {
    int p  = tid & (NT - 1);           // coalesced along pixels
    int c0 = (tid >> 7) * 2;           // channel pair start (384 threads -> 3 pairs/pass)
#pragma unroll
    for (int c = c0; c < DIMC; c += 2 * (TPB / NT)) {
      _Float16 e0 = (_Float16)xb[(size_t)c * HW + pt0 + p];
      _Float16 e1 = (_Float16)xb[(size_t)(c + 1) * HW + pt0 + p];
      union { _Float16 h[2]; unsigned u; } pk;
      pk.h[0] = e0; pk.h[1] = e1;
      *(unsigned*)(xs + p * SROW + c) = pk.u;
    }
  }
  __syncthreads();

  // ---- Stage 2: GEMM1 (384xNT = W_in * X) + bias + SimpleGate -> gs -------
  {
    v16h b = lds_frag(xs + ln * SROW + 8 * hi);   // prefetch nt=0, kf=0
#pragma unroll
    for (int nt = 0; nt < NSUB; ++nt) {
      v8f acc1 = bias1, acc2 = bias2;
      const _Float16* brow  = xs + (nt * 16 + ln) * SROW + 8 * hi;
      const _Float16* brow1 = xs + ((((nt + 1) & (NSUB - 1)) * 16) + ln) * SROW + 8 * hi;
#pragma unroll
      for (int kf = 0; kf < KF; ++kf) {
        v16h bn = (kf < KF - 1) ? lds_frag(brow + (kf + 1) * 32)
                                : lds_frag(brow1);       // wraps harmlessly at the end
        acc1 = __builtin_amdgcn_wmma_f32_16x16x32_f16(false, a1[kf], false, b,
                                                      (short)0, acc1, false, false);
        acc2 = __builtin_amdgcn_wmma_f32_16x16x32_f16(false, a2[kf], false, b,
                                                      (short)0, acc2, false, false);
        b = bn;
      }
      // gate in registers, f16 back to LDS (8 contiguous halves = 16B store)
      v8h gv;
#pragma unroll
      for (int r = 0; r < 8; ++r) gv[r] = (_Float16)(acc1[r] * acc2[r]);
      *(v8h*)(gs + (nt * 16 + ln) * SROW + wv * 16 + 8 * hi) = gv;
    }
  }
  __syncthreads();

  // ---- Stage 3: GEMM2 (192xNT = W_out * G) + bias -> global ---------------
  {
    // GEMM2 weights loaded here (not earlier) to keep stage-2 pressure low.
    v16h a3[KF];
#pragma unroll
    for (int kf = 0; kf < KF; ++kf) a3[kf] = A2[(wv * KF + kf) * 32 + lane];
    v8f bias3;
#pragma unroll
    for (int r = 0; r < 8; ++r) bias3[r] = b_out[wv * 16 + r + 8 * hi];

    v16h b = lds_frag(gs + ln * SROW + 8 * hi);   // prefetch nt=0, kf=0
#pragma unroll
    for (int nt = 0; nt < NSUB; ++nt) {
      v8f acc = bias3;
      const _Float16* brow  = gs + (nt * 16 + ln) * SROW + 8 * hi;
      const _Float16* brow1 = gs + ((((nt + 1) & (NSUB - 1)) * 16) + ln) * SROW + 8 * hi;
#pragma unroll
      for (int kf = 0; kf < KF; ++kf) {
        v16h bn = (kf < KF - 1) ? lds_frag(brow + (kf + 1) * 32)
                                : lds_frag(brow1);
        acc = __builtin_amdgcn_wmma_f32_16x16x32_f16(false, a3[kf], false, b,
                                                     (short)0, acc, false, false);
        b = bn;
      }
      int pcol = pt0 + nt * 16 + ln;
#pragma unroll
      for (int r = 0; r < 8; ++r) {
        ob[(size_t)(wv * 16 + r + 8 * hi) * HW + pcol] = acc[r];
      }
    }
  }
}

extern "C" void kernel_launch(void* const* d_in, const int* in_sizes, int n_in,
                              void* d_out, int out_size, void* d_ws, size_t ws_size,
                              hipStream_t stream) {
  const float* x     = (const float*)d_in[0];
  const float* w_in  = (const float*)d_in[1];
  const float* b_in  = (const float*)d_in[2];
  const float* w_out = (const float*)d_in[3];
  const float* b_out = (const float*)d_in[4];
  float* out = (float*)d_out;

  // workspace: packed f16 A-fragments
  _Float16* wA1 = (_Float16*)d_ws;                                   // 24*6*32*32B = 144KB
  _Float16* wA2 = (_Float16*)((char*)d_ws + (size_t)MT1 * KF * 32 * 32);

  {
    int n1 = MT1 * KF * 32;
    pack_weights_kernel<<<(n1 + 255) / 256, 256, 0, stream>>>(w_in, wA1, MT1);
    int n2 = MT2 * KF * 32;
    pack_weights_kernel<<<(n2 + 255) / 256, 256, 0, stream>>>(w_out, wA2, MT2);
  }

  int nblocks = 8 * (HW / NT);                 // 2304 tiles, all full
  size_t shbytes = (size_t)2 * NT * SROW * 2;  // xs + gs, f16: 102400 B
  simpleffn_kernel<<<nblocks, TPB, shbytes, stream>>>(x, wA1, b_in, wA2, b_out, out);
}
